// GCN_75909251990056
// MI455X (gfx1250) — matrix operand order repaired and checked by reading, not compiled, not run on hardware
//
#include <hip/hip_runtime.h>

#define NN 50000
#define NE 800000
#define D  128
#define DOUT 64
#define NG 50
#define EPSB 1e-5f

typedef __attribute__((ext_vector_type(16))) _Float16 v16h;
typedef __attribute__((ext_vector_type(8)))  float    v8f;

// ---------------------------------------------------------------------------
// Weight pre-pass: Wt[n*128 + k] = (f16) W[k*ncols + n]
// Makes each WMMA B-fragment one contiguous 32B load (kills the scalar
// load + s_wait_xcnt pointer-chase seen with strided f32 column loads).
// ---------------------------------------------------------------------------
__global__ void k_cvt_wt(const float* __restrict__ W, _Float16* __restrict__ Wt,
                         int ncols) {
    int idx = blockIdx.x * blockDim.x + threadIdx.x;   // < 128*ncols
    if (idx >= D * ncols) return;
    int k = idx / ncols, n = idx % ncols;
    Wt[(size_t)n * D + k] = (_Float16)W[idx];
}

// ---------------------------------------------------------------------------
// WMMA GEMM: H[nrows x ncols] = X[nrows x 128] @ W[128 x ncols]
// Block = 64 rows x 128 cols: 8 waves (one 16-col stripe each), 4 row tiles
// per wave sharing one B fragment per K-chunk. Wt is f16, K-major.
// A/B use contiguous K order (K = k0 + 16*half + i); C/D layout per CDNA5 ISA 7.12.2 (wave32).
// ---------------------------------------------------------------------------
__global__ __launch_bounds__(256)
void k_gemm_wmma(const float* __restrict__ X, const _Float16* __restrict__ Wt,
                 float* __restrict__ H, int nrows, int ncols) {
    const int lane = threadIdx.x & 31;
    const int wave = threadIdx.x >> 5;
    const int colBase = wave << 4;
    const int row0 = blockIdx.x * 64;
    if (row0 >= nrows || colBase >= ncols) return;

    const int m  = lane & 15;     // row within tile (A) / col within tile (B,C)
    const int hi = lane >> 4;     // lane half selects K sub-group
    const int n  = colBase + m;
    int nt = (nrows - row0 + 15) >> 4;   // valid 16-row tiles in this block
    if (nt > 4) nt = 4;

    const float* __restrict__ x0 = X + (size_t)(row0 + m) * D;
    v8f acc0 = {}, acc1 = {}, acc2 = {}, acc3 = {};

#pragma unroll
    for (int kt = 0; kt < 4; ++kt) {
        const int k0 = kt * 32;
        // B 32x16 f16: lane -> N = colBase+m, K = k0 + hi*16 + h (contiguous)
        v16h b = *(const v16h*)(Wt + (size_t)n * D + k0 + hi * 16);
        const int ka = k0 + hi * 16;  // A: halves 0..15 -> K=ka+h (contiguous)

#define GCN_TILE(t, accv)                                                   \
        if (nt > (t)) {                                                     \
            const float* __restrict__ xr = x0 + (size_t)((t) * 16) * D;     \
            v16h a;                                                         \
            _Pragma("unroll")                                               \
            for (int h = 0; h < 8; ++h) {                                   \
                a[h]     = (_Float16)xr[ka + h];                            \
                a[h + 8] = (_Float16)xr[ka + 8 + h];                        \
            }                                                               \
            accv = __builtin_amdgcn_wmma_f32_16x16x32_f16(                  \
                false, a, false, b, (short)0, accv, false, false);          \
        }
        GCN_TILE(0, acc0)
        GCN_TILE(1, acc1)
        GCN_TILE(2, acc2)
        GCN_TILE(3, acc3)
#undef GCN_TILE
    }

    // C/D f32 16x16: VGPR r -> M = r + hi*8, N = colBase + m
#define GCN_STORE(t, accv)                                                  \
    if (nt > (t)) {                                                         \
        const int rb = row0 + (t) * 16 + hi * 8;                            \
        _Pragma("unroll")                                                   \
        for (int r = 0; r < 8; ++r)                                         \
            H[(size_t)(rb + r) * ncols + n] = accv[r];                      \
    }
    GCN_STORE(0, acc0)
    GCN_STORE(1, acc1)
    GCN_STORE(2, acc2)
    GCN_STORE(3, acc3)
#undef GCN_STORE
}

// ---------------------------------------------------------------------------
// Degree / normalization helpers
// ---------------------------------------------------------------------------
__global__ void k_fill1(float* p, int n) {
    int i = blockIdx.x * blockDim.x + threadIdx.x;
    if (i < n) p[i] = 1.0f;                       // self-loop contribution
}
__global__ void k_deg_accum(const int* __restrict__ col, float* deg, int ne) {
    int i = blockIdx.x * blockDim.x + threadIdx.x;
    if (i < ne) atomicAdd(&deg[col[i]], 1.0f);
}
__global__ void k_rsqrt(float* p, int n) {
    int i = blockIdx.x * blockDim.x + threadIdx.x;
    if (i < n) { float d = p[i]; p[i] = d > 0.f ? rsqrtf(d) : 0.f; }
}

// out[i,f] = dinv[i]^2 * h[i,f] + bias[f]   (self-loop term + conv bias)
__global__ __launch_bounds__(256)
void k_agg_init(const float* __restrict__ h, const float* __restrict__ dinv,
                const float* __restrict__ bias, float* __restrict__ out) {
    int idx = blockIdx.x * blockDim.x + threadIdx.x;   // < NN*D
    int node = idx >> 7, f = idx & 127;
    float di = dinv[node];
    out[idx] = di * di * h[idx] + bias[f];
}

// one wave per edge: 32 lanes x float4 covers the 128-feature row (coalesced)
__global__ __launch_bounds__(256)
void k_agg_edges(const float* __restrict__ h, const int* __restrict__ row,
                 const int* __restrict__ col, const float* __restrict__ dinv,
                 float* __restrict__ out, int ne) {
    long long idx = (long long)blockIdx.x * blockDim.x + threadIdx.x;
    int e = (int)(idx >> 5);
    if (e >= ne) return;
    int g = (int)(idx & 31);
    int r = row[e], c = col[e];
    if (e + 64 < ne)   // warm L2/WGP$ for an upcoming source row -> global_prefetch_b8
        __builtin_prefetch(h + (size_t)row[e + 64] * D, 0, 0);
    float nm = dinv[r] * dinv[c];
    float4 v = ((const float4*)(h + (size_t)r * D))[g];
    float* o = out + (size_t)c * D + g * 4;
    atomicAdd(o + 0, v.x * nm);
    atomicAdd(o + 1, v.y * nm);
    atomicAdd(o + 2, v.z * nm);
    atomicAdd(o + 3, v.w * nm);
}

__global__ void k_relu(float* p, int n) {
    int i = blockIdx.x * blockDim.x + threadIdx.x;
    if (i < n) p[i] = fmaxf(p[i], 0.f);
}

// per-feature sum & sumsq (row stride fixed at D); one block per feature
__global__ __launch_bounds__(256)
void k_bn_stats(const float* __restrict__ X, float* __restrict__ stats, int nrows) {
    __shared__ float ss[256], s2[256];
    int f = blockIdx.x;
    float a = 0.f, b = 0.f;
    for (int r = threadIdx.x; r < nrows; r += 256) {
        float v = X[(size_t)r * D + f];
        a += v; b += v * v;
    }
    ss[threadIdx.x] = a; s2[threadIdx.x] = b;
    __syncthreads();
    for (int s = 128; s > 0; s >>= 1) {
        if (threadIdx.x < s) {
            ss[threadIdx.x] += ss[threadIdx.x + s];
            s2[threadIdx.x] += s2[threadIdx.x + s];
        }
        __syncthreads();
    }
    if (threadIdx.x == 0) { stats[f] = ss[0]; stats[D + f] = s2[0]; }
}

// x = relu((x - m) * rsqrt(v + eps) * gamma + beta); biased variance
__global__ __launch_bounds__(256)
void k_bn_apply_relu(float* __restrict__ X, const float* __restrict__ stats,
                     const float* __restrict__ gamma, const float* __restrict__ beta,
                     int nrows) {
    int idx = blockIdx.x * blockDim.x + threadIdx.x;
    if (idx >= nrows * D) return;
    int f = idx & 127;
    float inv_n = 1.f / (float)nrows;
    float m = stats[f] * inv_n;
    float v = stats[D + f] * inv_n - m * m;
    float y = (X[idx] - m) * rsqrtf(v + EPSB) * gamma[f] + beta[f];
    X[idx] = fmaxf(y, 0.f);
}

// ---------------------------------------------------------------------------
// Mean pool per graph
// ---------------------------------------------------------------------------
__global__ void k_zero(float* p, int n) {
    int i = blockIdx.x * blockDim.x + threadIdx.x;
    if (i < n) p[i] = 0.f;
}
__global__ void k_pool_cnt(const int* __restrict__ batch, float* cnts, int n) {
    int i = blockIdx.x * blockDim.x + threadIdx.x;
    if (i < n) atomicAdd(&cnts[batch[i]], 1.0f);
}
__global__ __launch_bounds__(256)
void k_pool_accum(const float* __restrict__ X, const int* __restrict__ batch,
                  float* __restrict__ sums) {
    long long idx = (long long)blockIdx.x * blockDim.x + threadIdx.x;  // < NN*32
    int node = (int)(idx >> 5), g4 = (int)(idx & 31);
    int gr = batch[node];
    float4 v = ((const float4*)(X + (size_t)node * D))[g4];
    float* o = sums + (size_t)gr * D + g4 * 4;
    atomicAdd(o + 0, v.x); atomicAdd(o + 1, v.y);
    atomicAdd(o + 2, v.z); atomicAdd(o + 3, v.w);
}
// pooled means zero-padded to 64 rows so the WMMA GEMM can consume them
__global__ void k_pool_div_pad(const float* __restrict__ sums,
                               const float* __restrict__ cnts,
                               float* __restrict__ pad) {
    int idx = blockIdx.x * blockDim.x + threadIdx.x;   // < 64*D
    int g = idx >> 7, f = idx & 127;
    pad[idx] = (g < NG) ? sums[g * D + f] / fmaxf(cnts[g], 1.0f) : 0.f;
}
__global__ void k_bias_add(float* __restrict__ y, const float* __restrict__ b, int n) {
    int i = blockIdx.x * blockDim.x + threadIdx.x;
    if (i < n) y[i] += b[i & 127];
}
__global__ void k_final(const float* __restrict__ y1, const float* __restrict__ b,
                        float* __restrict__ out) {
    int idx = blockIdx.x * blockDim.x + threadIdx.x;   // < NG*DOUT
    if (idx >= NG * DOUT) return;
    int f = idx % DOUT;
    out[idx] = fmaxf(y1[idx] + b[f], 0.f);
}

// ---------------------------------------------------------------------------
extern "C" void kernel_launch(void* const* d_in, const int* in_sizes, int n_in,
                              void* d_out, int out_size, void* d_ws, size_t ws_size,
                              hipStream_t stream) {
    (void)in_sizes; (void)n_in; (void)out_size; (void)ws_size;
    const float* x     = (const float*)d_in[0];
    const int*   ei    = (const int*)d_in[1];
    const int*   batch = (const int*)d_in[2];
    const float* Wc[4] = {(const float*)d_in[3], (const float*)d_in[5],
                          (const float*)d_in[7], (const float*)d_in[9]};
    const float* bc[4] = {(const float*)d_in[4], (const float*)d_in[6],
                          (const float*)d_in[8], (const float*)d_in[10]};
    const float* gn_g = (const float*)d_in[11];
    const float* gn_b = (const float*)d_in[12];
    const float* l0W  = (const float*)d_in[13];
    const float* l0b  = (const float*)d_in[14];
    const float* l1W  = (const float*)d_in[15];
    const float* l1b  = (const float*)d_in[16];
    const float* bn_g = (const float*)d_in[17];
    const float* bn_b = (const float*)d_in[18];
    const int* row = ei;            // sources
    const int* col = ei + NE;       // targets

    float* ws    = (float*)d_ws;
    float* h     = ws;                                   // NN*D
    float* buf0  = h    + (size_t)NN * D;                // NN*D
    float* buf1  = buf0 + (size_t)NN * D;                // NN*D
    float* dinv  = buf1 + (size_t)NN * D;                // NN
    float* stats = dinv + NN;                            // 2*D
    float* sums  = stats + 2 * D;                        // NG*D
    float* cnts  = sums + NG * D;                        // 64 (pad)
    float* pad   = cnts + 64;                            // 64*D
    float* y0    = pad  + 64 * D;                        // 64*D
    float* y1    = y0   + 64 * D;                        // 64*DOUT
    _Float16* wt = (_Float16*)(y1 + 64 * DOUT);          // f16 transposed weights
    _Float16* Wt[4] = {wt, wt + D * D, wt + 2 * D * D, wt + 3 * D * D};
    _Float16* l0Wt  = wt + 4 * (size_t)D * D;            // 128x128
    _Float16* l1Wt  = wt + 5 * (size_t)D * D;            // 64x128

    // --- weight transpose+f16 pre-pass (contiguous WMMA B fragments) ---
    for (int l = 0; l < 4; ++l)
        k_cvt_wt<<<(D * D) / 256, 256, 0, stream>>>(Wc[l], Wt[l], D);
    k_cvt_wt<<<(D * D) / 256, 256, 0, stream>>>(l0W, l0Wt, D);
    k_cvt_wt<<<(D * DOUT) / 256, 256, 0, stream>>>(l1W, l1Wt, DOUT);

    // --- symmetric GCN normalization: dinv = rsqrt(1 + in-degree) ---
    k_fill1<<<(NN + 255) / 256, 256, 0, stream>>>(dinv, NN);
    k_deg_accum<<<(NE + 255) / 256, 256, 0, stream>>>(col, dinv, NE);
    k_rsqrt<<<(NN + 255) / 256, 256, 0, stream>>>(dinv, NN);

    // --- 4 GCN conv layers ---
    const float* cur = x;
    float* bufs[2] = {buf0, buf1};
    for (int l = 0; l < 4; ++l) {
        float* nxt = bufs[l & 1];
        k_gemm_wmma<<<(NN + 63) / 64, 256, 0, stream>>>(cur, Wt[l], h, NN, D);
        k_agg_init<<<(NN * D) / 256, 256, 0, stream>>>(h, dinv, bc[l], nxt);
        k_agg_edges<<<(int)(((long long)NE * 32) / 256), 256, 0, stream>>>(
            h, row, col, dinv, nxt, NE);
        if (l == 2) {
            k_bn_stats<<<D, 256, 0, stream>>>(nxt, stats, NN);
            k_bn_apply_relu<<<(NN * D) / 256, 256, 0, stream>>>(nxt, stats, gn_g, gn_b, NN);
        } else {
            k_relu<<<(NN * D) / 256, 256, 0, stream>>>(nxt, NN * D);
        }
        cur = nxt;
    }

    // --- mean pool per graph ---
    k_zero<<<(NG * D + 64 + 255) / 256, 256, 0, stream>>>(sums, NG * D + 64);
    k_pool_cnt<<<(NN + 255) / 256, 256, 0, stream>>>(batch, cnts, NN);
    k_pool_accum<<<(NN * 32) / 256, 256, 0, stream>>>(cur, batch, sums);
    k_pool_div_pad<<<(64 * D) / 256, 256, 0, stream>>>(sums, cnts, pad);

    // --- MLP head (reuses WMMA GEMM on 64-row padded tiles, single block) ---
    k_gemm_wmma<<<1, 256, 0, stream>>>(pad, l0Wt, y0, 64, D);
    k_bias_add<<<(NG * D + 255) / 256, 256, 0, stream>>>(y0, l0b, NG * D);
    k_bn_stats<<<D, 256, 0, stream>>>(y0, stats, NG);
    k_bn_apply_relu<<<(NG * D + 255) / 256, 256, 0, stream>>>(y0, stats, bn_g, bn_b, NG);

    k_gemm_wmma<<<1, 256, 0, stream>>>(y0, l1Wt, y1, 64, DOUT);
    k_final<<<(NG * DOUT + 255) / 256, 256, 0, stream>>>(y1, l1b, (float*)d_out);
}
